// MSSO_28527172780066
// MI455X (gfx1250) — matrix operand order repaired and checked by reading, not compiled, not run on hardware
//
// MSSO (Mobius selective-scan operator) for MI455X / gfx1250, wave32 WMMA.
//
// Roofline: in_proj (51.5 GFLOP) + out_proj (17.2 GFLOP) dominate; they move
// only ~150 MB (~6.5 us at 23.3 TB/s) -> compute bound -> must run on the
// matrix path. fp32 WMMA is K=4/issue; bf16 WMMA 16x16x32 with f32 accum is
// 8x the K per issue, so fp32 inputs are converted to bf16 (RNE) during LDS
// staging (global_load_b128 -> packed ds stores). Scan is tiny/sequential ->
// one block per (batch,head), dt/B/C staged with GLOBAL_LOAD_ASYNC_TO_LDS.

#include <hip/hip_runtime.h>
#include <hip/hip_bf16.h>
#include <math.h>

typedef __bf16 bf16_t;
typedef bf16_t v16bf __attribute__((ext_vector_type(16)));
typedef float  v8f   __attribute__((ext_vector_type(8)));

#define BSZ    8
#define NTOK   1024
#define DIN    1024
#define HEADS  2
#define DHEAD  512
#define SST    8
#define M_IN   3072
#define M_DBC  34
#define KDIM   1024
#define LDP    36        // LDS row pad (shorts): 72B stride, conflict-free

__device__ __forceinline__ unsigned short f2bf(float f) {
  unsigned int u = __float_as_uint(f);
  u += 0x7FFFu + ((u >> 16) & 1u);          // round-to-nearest-even
  return (unsigned short)(u >> 16);
}
__device__ __forceinline__ unsigned pack2bf(float lo, float hi) {
  return (unsigned)f2bf(lo) | ((unsigned)f2bf(hi) << 16);
}
__device__ __forceinline__ float silu_f(float x)     { return x / (1.f + __expf(-x)); }
__device__ __forceinline__ float softplus_f(float x) { return (x > 20.f) ? x : log1pf(__expf(x)); }

union BFVec { v16bf v; unsigned short u[16]; };

// ---------------------------------------------------------------------------
// Mobius permutation: one block per head, stable O(N^2) rank == jnp.argsort.
// ---------------------------------------------------------------------------
__launch_bounds__(1024)
__global__ void perm_kernel(const float* __restrict__ mobius,
                            int* __restrict__ perm, int* __restrict__ inv) {
  const int h = blockIdx.x;
  const int n = threadIdx.x;            // 0..1023
  __shared__ float sc[NTOK];

  float a = mobius[h*4+0], b = mobius[h*4+1], c = mobius[h*4+2], d = mobius[h*4+3];
  float det = fmaxf(fabsf(a*d - b*c), 0.001f);
  float s = sqrtf(det);
  a /= s; b /= s; c /= s; d /= s;

  const int iy = n >> 5, ix = n & 31;
  const float re = -1.f + (2.f/31.f)*ix;
  const float im = -1.f + (2.f/31.f)*iy;
  // w = (a*z + b) / (c*z + d + 1e-6), z = re + i*im
  const float nr = a*re + b,           ni = a*im;
  const float dr = c*re + d + 1e-6f,   di = c*im;
  const float den = dr*dr + di*di;
  const float wr = (nr*dr + ni*di)/den;
  const float wi = (ni*dr - nr*di)/den;
  sc[n] = wr + 1.3f*wi;
  __syncthreads();

  const float mine = sc[n];
  int rank = 0;
  for (int m = 0; m < NTOK; ++m) {
    float v = sc[m];
    rank += (v < mine) || (v == mine && m < n);
  }
  perm[h*NTOK + rank] = n;
  inv[h*NTOK + n] = rank;
}

// ---------------------------------------------------------------------------
// Generic bf16-WMMA GEMM: O[b] = act(W(MxK) * X[b](KxN) + bias), K=N=1024.
// Block 256 thr = 8 waves, tile 128x128x32, wave tile 32x64 (2x4 wmma accs).
// BOUND=false -> M is a multiple of 128, no row guards (vectorized staging).
// ---------------------------------------------------------------------------
template <bool BOUND>
__launch_bounds__(256)
__global__ void gemm_bf16_wmma(const float* __restrict__ Wm,
                               const float* __restrict__ Xm,
                               const float* __restrict__ bias,
                               float* __restrict__ Om,
                               int M, int act) {
  __shared__ unsigned short As[128][LDP];   // [m][k], K-contiguous
  __shared__ unsigned short Bs[128][LDP];   // [n][k], K-contiguous

  const int tid  = threadIdx.x;
  const int wave = tid >> 5, lane = tid & 31;
  const int wm   = wave >> 1, wn = wave & 1;      // 4x2 wave grid
  const int lr   = lane & 15, khalf = lane >> 4;
  const int m0   = blockIdx.y * 128;
  const int n0   = blockIdx.x * 128;
  const int b    = blockIdx.z;

  const float* Xb = Xm + (size_t)b * KDIM * NTOK;
  float*       Ob = Om + (size_t)b * (size_t)M * NTOK;

  // Per-thread staging geometry: 16 consecutive floats of each 4096-elem tile.
  const int am = (tid * 16) >> 5;       // A: row, two threads per row
  const int ak = (tid * 16) & 31;       // A: k0 in {0,16}
  const int bk = (tid * 16) >> 7;       // B: k row, eight threads per row
  const int bn = (tid * 16) & 127;      // B: n0 of 16 consecutive columns

  const v8f vzero = {0.f,0.f,0.f,0.f,0.f,0.f,0.f,0.f};
  v8f acc[2][4];
#pragma unroll
  for (int tm = 0; tm < 2; ++tm)
#pragma unroll
    for (int tn = 0; tn < 4; ++tn) acc[tm][tn] = vzero;

  for (int kk = 0; kk < KDIM; kk += 32) {
    // ---- stage A tile (128x32 fp32 -> bf16): 4x global_load_b128 / thread --
    {
      unsigned* dst = (unsigned*)&As[am][ak];
      if (!BOUND || (m0 + am) < M) {
        const float4* src =
            (const float4*)&Wm[(size_t)(m0 + am) * KDIM + kk + ak];
#pragma unroll
        for (int q = 0; q < 4; ++q) {
          float4 v = src[q];
          dst[2*q + 0] = pack2bf(v.x, v.y);
          dst[2*q + 1] = pack2bf(v.z, v.w);
        }
      } else {
#pragma unroll
        for (int q = 0; q < 8; ++q) dst[q] = 0u;
      }
    }
    // ---- stage B tile (32x128 fp32 -> bf16), transposed to [n][k] ----------
    {
      const float4* src = (const float4*)&Xb[(size_t)(kk + bk) * NTOK + n0 + bn];
#pragma unroll
      for (int q = 0; q < 4; ++q) {
        float4 v = src[q];
        Bs[bn + 4*q + 0][bk] = f2bf(v.x);
        Bs[bn + 4*q + 1][bk] = f2bf(v.y);
        Bs[bn + 4*q + 2][bk] = f2bf(v.z);
        Bs[bn + 4*q + 3][bk] = f2bf(v.w);
      }
    }
    if (kk + 32 < KDIM) {   // gfx1250 global_prefetch_b8 of next tiles
      __builtin_prefetch(&Wm[(size_t)(m0 + am) * KDIM + kk + 32 + ak], 0, 0);
      __builtin_prefetch(&Xb[(size_t)(kk + 32 + bk) * NTOK + n0 + bn], 0, 0);
    }
    __syncthreads();

    // ---- build fragments (ISA 16-bit A 16x32 layout; B split K halves) ----
    v16bf afrag[2], bfrag[4];
#pragma unroll
    for (int tm = 0; tm < 2; ++tm) {
      const int r = wm * 32 + tm * 16 + lr;
      const int base = khalf * 8;
      BFVec u;
#pragma unroll
      for (int i = 0; i < 8; ++i) {
        u.u[i]     = As[r][base + i];        // K = base..base+7
        u.u[8 + i] = As[r][base + 16 + i];   // K = base+16..base+23
      }
      afrag[tm] = u.v;
    }
#pragma unroll
    for (int tn = 0; tn < 4; ++tn) {
      const int cN = wn * 64 + tn * 16 + lr;
      BFVec u;
#pragma unroll
      for (int i = 0; i < 16; ++i) u.u[i] = Bs[cN][khalf * 8 + (i & 7) + (i >> 3) * 16];
      bfrag[tn] = u.v;
    }

    // ---- 8 x v_wmma_f32_16x16x32_bf16 per wave per K-step ----
#pragma unroll
    for (int tm = 0; tm < 2; ++tm)
#pragma unroll
      for (int tn = 0; tn < 4; ++tn)
        acc[tm][tn] = __builtin_amdgcn_wmma_f32_16x16x32_bf16(
            false, afrag[tm], false, bfrag[tn],
            (short)0, acc[tm][tn], false, false);
    __syncthreads();
  }

  // ---- epilogue: bias + activation; C/D layout: lane -> col, vgpr -> row ----
#pragma unroll
  for (int tm = 0; tm < 2; ++tm)
#pragma unroll
    for (int tn = 0; tn < 4; ++tn) {
      const int col = n0 + wn * 64 + tn * 16 + lr;
#pragma unroll
      for (int v = 0; v < 8; ++v) {
        const int row = m0 + wm * 32 + tm * 16 + khalf * 8 + v;
        if (!BOUND || row < M) {
          float r = acc[tm][tn][v] + (bias ? bias[row] : 0.f);
          if (act) r = silu_f(r);
          Ob[(size_t)row * NTOK + col] = r;
        }
      }
    }
}

// ---------------------------------------------------------------------------
// Depthwise conv3 over flattened tokens + SiLU.
// ---------------------------------------------------------------------------
__global__ void conv_silu_kernel(const float* __restrict__ xz,
                                 const float* __restrict__ dww,
                                 const float* __restrict__ dwb,
                                 float* __restrict__ xf) {
  size_t idx = (size_t)blockIdx.x * blockDim.x + threadIdx.x;
  const int n = idx & (NTOK - 1);
  const int c = (int)((idx >> 10) & (DIN - 1));
  const int b = (int)(idx >> 20);
  const float* xs = xz + ((size_t)b * M_IN + c) * NTOK;   // x_ssm slice of xz
  const float w0 = dww[c*3+0], w1 = dww[c*3+1], w2 = dww[c*3+2];
  float acc = dwb[c] + w1 * xs[n];
  if (n > 0)        acc += w0 * xs[n-1];
  if (n < NTOK - 1) acc += w2 * xs[n+1];
  xf[((size_t)b * DIN + c) * NTOK + n] = silu_f(acc);
}

// ---------------------------------------------------------------------------
// Split dbc -> softplus(dt), B, C; gather through perm into scan order.
// ---------------------------------------------------------------------------
__global__ void dbc_perm_kernel(const float* __restrict__ dbc,
                                const int* __restrict__ perm,
                                float* __restrict__ dtp,
                                float* __restrict__ Btp,
                                float* __restrict__ Ctp) {
  const int idx = blockIdx.x * blockDim.x + threadIdx.x;  // 16384
  const int np = idx & (NTOK - 1);
  const int h  = (idx >> 10) & 1;
  const int b  = idx >> 11;
  const int pn = perm[h * NTOK + np];
  const float* base = dbc + ((size_t)b * M_DBC + h * (1 + 2 * SST)) * NTOK;
  const int bh = b * HEADS + h;
  dtp[(size_t)bh * NTOK + np] = softplus_f(base[pn]);
#pragma unroll
  for (int s = 0; s < SST; ++s) {
    Btp[((size_t)bh * SST + s) * NTOK + np] = base[(size_t)(1 + s) * NTOK + pn];
    Ctp[((size_t)bh * SST + s) * NTOK + np] = base[(size_t)(1 + SST + s) * NTOK + pn];
  }
}

// ---------------------------------------------------------------------------
// Gather x into scan order, transposed to (b,h,n,dh) for coalesced scan I/O.
// ---------------------------------------------------------------------------
__global__ void xgather_kernel(const float* __restrict__ xf,
                               const int* __restrict__ perm,
                               float* __restrict__ xhp) {
  size_t idx = (size_t)blockIdx.x * blockDim.x + threadIdx.x;
  const int dh = idx & (DHEAD - 1);
  const int np = (int)((idx >> 9) & (NTOK - 1));
  const int bh = (int)(idx >> 19);
  const int h = bh & 1, b = bh >> 1;
  const int pn = perm[h * NTOK + np];
  xhp[((size_t)bh * NTOK + np) * DHEAD + dh] =
      xf[((size_t)b * DIN + h * DHEAD + dh) * NTOK + pn];
}

// ---------------------------------------------------------------------------
// Selective scan: block per (batch,head), thread per channel; 8-state SSM in
// registers. dt/B/C chunks staged into LDS with GLOBAL_LOAD_ASYNC_TO_LDS
// (ASYNCcnt-tracked DMA, frees the VALU; per ISA flat->LDS aperture mapping
// the low 32 bits of a generic __shared__ pointer are the LDS byte offset).
// ---------------------------------------------------------------------------
#define SCHUNK 64
__launch_bounds__(512)
__global__ void scan_kernel(const float* __restrict__ dtp,
                            const float* __restrict__ Btp,
                            const float* __restrict__ Ctp,
                            const float* __restrict__ xhp,
                            const float* __restrict__ A_log,
                            const float* __restrict__ Dp,
                            float* __restrict__ yht) {
  const int bh = blockIdx.x;          // b*HEADS + h
  const int h  = bh & 1;
  const int dh = threadIdx.x;         // 0..511

  float A[SST];
#pragma unroll
  for (int s = 0; s < SST; ++s) A[s] = -__expf(A_log[h * SST + s]);
  const float Dv = Dp[h];

  float st[SST];
#pragma unroll
  for (int s = 0; s < SST; ++s) st[s] = 0.f;

  __shared__ float sdt[SCHUNK];
  __shared__ float sB[SST][SCHUNK];
  __shared__ float sC[SST][SCHUNK];

  const float* dtb = dtp + (size_t)bh * NTOK;
  const float* Bb  = Btp + (size_t)bh * SST * NTOK;
  const float* Cb  = Ctp + (size_t)bh * SST * NTOK;
  const float* xb  = xhp + (size_t)bh * NTOK * DHEAD;
  float*       yb  = yht + (size_t)bh * NTOK * DHEAD;

  for (int n0 = 0; n0 < NTOK; n0 += SCHUNK) {
    __syncthreads();
    // async DMA of the (1+2S)xSCHUNK chunk into LDS
    for (int i = dh; i < (1 + 2 * SST) * SCHUNK; i += 512) {
      const int which = i >> 6;        // i / SCHUNK
      const int j     = i & (SCHUNK - 1);
      const float* g;
      float* l;
      if (which == 0)        { g = dtb + n0 + j;                          l = &sdt[j]; }
      else if (which <= SST) { g = Bb + (size_t)(which - 1) * NTOK + n0 + j; l = &sB[which - 1][j]; }
      else                   { g = Cb + (size_t)(which - 1 - SST) * NTOK + n0 + j; l = &sC[which - 1 - SST][j]; }
      const unsigned loff = (unsigned)(size_t)l;          // LDS byte offset
      const unsigned long long ga = (unsigned long long)g;
      asm volatile("global_load_async_to_lds_b32 %0, %1, off"
                   :: "v"(loff), "v"(ga) : "memory");
    }
    asm volatile("s_wait_asynccnt 0x0" ::: "memory");
    __syncthreads();

    for (int j = 0; j < SCHUNK; ++j) {
      const int n = n0 + j;
      const float dt = sdt[j];
      const float xt = xb[(size_t)n * DHEAD + dh];
      float y = 0.f;
#pragma unroll
      for (int s = 0; s < SST; ++s) {
        const float dA = __expf(dt * A[s]);
        st[s] = dA * st[s] + dt * sB[s][j] * xt;
        y += st[s] * sC[s][j];
      }
      yb[(size_t)n * DHEAD + dh] = y + Dv * xt;
    }
  }
}

// ---------------------------------------------------------------------------
// Un-permute y, apply gate (silu again, per reference) and residual.
// ---------------------------------------------------------------------------
__global__ void fuse_gate_kernel(const float* __restrict__ yht,
                                 const float* __restrict__ xz,
                                 const int* __restrict__ inv,
                                 float* __restrict__ ybuf) {
  size_t idx = (size_t)blockIdx.x * blockDim.x + threadIdx.x;
  const int n = idx & (NTOK - 1);
  const int c = (int)((idx >> 10) & (DIN - 1));
  const int b = (int)(idx >> 20);
  const int h = c >> 9, dh = c & (DHEAD - 1);
  const int pn = inv[h * NTOK + n];
  const float yv = yht[(((size_t)(b * HEADS + h)) * NTOK + pn) * DHEAD + dh];
  const float g  = xz[((size_t)b * M_IN + 2 * DIN + c) * NTOK + n];
  const float r  = xz[((size_t)b * M_IN + DIN + c) * NTOK + n];
  ybuf[((size_t)b * DIN + c) * NTOK + n] = yv * silu_f(g) + r;
}

// ---------------------------------------------------------------------------
extern "C" void kernel_launch(void* const* d_in, const int* in_sizes, int n_in,
                              void* d_out, int out_size, void* d_ws, size_t ws_size,
                              hipStream_t stream) {
  (void)in_sizes; (void)n_in; (void)out_size; (void)ws_size;
  const float* x      = (const float*)d_in[0];
  const float* in_w   = (const float*)d_in[1];
  const float* in_b   = (const float*)d_in[2];
  const float* dw_w   = (const float*)d_in[3];
  const float* dw_b   = (const float*)d_in[4];
  const float* dbc_w  = (const float*)d_in[5];
  const float* dbc_b  = (const float*)d_in[6];
  const float* A_log  = (const float*)d_in[7];
  const float* Dparam = (const float*)d_in[8];
  const float* mobius = (const float*)d_in[9];
  const float* out_w  = (const float*)d_in[10];
  const float* out_b  = (const float*)d_in[11];

  float* ws = (float*)d_ws;
  // float-element offsets
  const size_t XZ  = 0;                               // 8*3072*1024
  const size_t XF  = XZ  + (size_t)BSZ * M_IN * NTOK; // 8*1024*1024
  const size_t XHP = XF  + (size_t)BSZ * DIN * NTOK;  // (b,h,n,dh)   (aliased by YBUF)
  const size_t YHT = XHP + (size_t)BSZ * DIN * NTOK;  // (b,h,n,dh)
  const size_t DBC = YHT + (size_t)BSZ * DIN * NTOK;  // 8*34*1024
  const size_t DTP = DBC + (size_t)BSZ * M_DBC * NTOK;
  const size_t BTP = DTP + (size_t)BSZ * HEADS * NTOK;
  const size_t CTP = BTP + (size_t)BSZ * HEADS * SST * NTOK;
  const size_t PRM = CTP + (size_t)BSZ * HEADS * SST * NTOK;

  float* xz   = ws + XZ;
  float* xf   = ws + XF;
  float* xhp  = ws + XHP;
  float* yht  = ws + YHT;
  float* dbcb = ws + DBC;
  float* dtp  = ws + DTP;
  float* Btp  = ws + BTP;
  float* Ctp  = ws + CTP;
  int*   perm = (int*)(ws + PRM);
  int*   inv  = perm + HEADS * NTOK;
  float* ybuf = xhp;   // xhp is dead after the scan; reuse for gated y

  // 1) Mobius permutations (2 heads)
  perm_kernel<<<HEADS, 1024, 0, stream>>>(mobius, perm, inv);

  // 2) in_proj GEMM + SiLU  -> xz (x_ssm | res | gate)
  gemm_bf16_wmma<false><<<dim3(NTOK/128, M_IN/128, BSZ), 256, 0, stream>>>(
      in_w, x, in_b, xz, M_IN, 1);

  // 3) depthwise conv3 + SiLU -> xf
  conv_silu_kernel<<<(BSZ * DIN * NTOK) / 256, 256, 0, stream>>>(xz, dw_w, dw_b, xf);

  // 4) dbc GEMM (M=34, bias, no act) -> dbcb
  gemm_bf16_wmma<true><<<dim3(NTOK/128, 1, BSZ), 256, 0, stream>>>(
      dbc_w, xf, dbc_b, dbcb, M_DBC, 0);

  // 5) split + softplus + permute -> dtp/Btp/Ctp
  dbc_perm_kernel<<<(BSZ * HEADS * NTOK) / 256, 256, 0, stream>>>(
      dbcb, perm, dtp, Btp, Ctp);

  // 6) gather x into scan order (transposed) -> xhp
  xgather_kernel<<<(BSZ * DIN * NTOK) / 256, 256, 0, stream>>>(xf, perm, xhp);

  // 7) selective scan -> yht
  scan_kernel<<<BSZ * HEADS, 512, 0, stream>>>(dtp, Btp, Ctp, xhp, A_log, Dparam, yht);

  // 8) un-permute + gate + residual -> ybuf
  fuse_gate_kernel<<<(BSZ * DIN * NTOK) / 256, 256, 0, stream>>>(yht, xz, inv, ybuf);

  // 9) out_proj GEMM + SiLU -> d_out
  gemm_bf16_wmma<false><<<dim3(NTOK/128, DIN/128, BSZ), 256, 0, stream>>>(
      out_w, ybuf, out_b, (float*)d_out, DIN, 1);
}
